// Decoder_13048110645533
// MI455X (gfx1250) — matrix operand order repaired and checked by reading, not compiled, not run on hardware
//
#include <hip/hip_runtime.h>

// ---------------- types / helpers ----------------
typedef __attribute__((ext_vector_type(16))) __bf16        v16bf;
typedef __attribute__((ext_vector_type(8)))  float         v8f;
typedef __attribute__((ext_vector_type(4)))  unsigned int  v4u;

union ABFrag { v16bf v; v4u q[2]; };

static __device__ __forceinline__ unsigned short f2bf(float f) {
  unsigned int u = __float_as_uint(f);
  u += 0x7FFFu + ((u >> 16) & 1u);       // round-to-nearest-even
  return (unsigned short)(u >> 16);
}

// ---------------- problem dims ----------------
constexpr int V_  = 32000, E_ = 256, EH_ = 1024, H_ = 512;
constexpr int B_  = 32, T_ = 64, S_ = 128, TS_ = 63;   // TS = T-1 serial steps
constexpr int KU_ = EH_ + H_;   // 1536 : per-step K for LSTM and vocab GEMMs
constexpr int G4_ = 4 * H_;     // 2048 : LSTM gate width

// =====================================================================
// Generic WMMA GEMM:  D[m,n] (+)= sum_k A[m,k] * W[n,k]  (+bias +addend)
// A, W bf16 (ushort storage) row-major; D fp32.  M must be a multiple of
// 32, N a multiple of 256, K a multiple of 32.
// Block = 256 threads = 8 waves. Each wave owns BOTH 16-row m-tiles and
// TWO 16-col n-tiles (2x2 tiles, 4 accumulators): each B fragment feeds
// two WMMAs and each A fragment feeds two WMMAs -> weights are read from
// L2 exactly once, A LDS reads halved.  Block covers 32 x 256 output.
// A panel (32 x K) staged in LDS via async global->LDS (CDNA5 path),
// row pitch K+8 to avoid bank conflicts.
// OUT_MAP: output row = (r&31)*T + (r>>5) + tplus1  (scatter into [B,T,V])
// ACCUM  : D += acc (serial vocab projection accumulates onto pre_logits)
// HAS_ADD: acc += addend[r*ldadd + n] (pre_gates)
// =====================================================================
template<bool OUT_MAP, bool ACCUM, bool HAS_ADD>
__global__ __launch_bounds__(256) void gemm_wmma(
    const unsigned short* __restrict__ A, int lda,
    const unsigned short* __restrict__ W, int ldw,
    float* __restrict__ D, int ldo,
    const float* __restrict__ bias, const float* __restrict__ bias2,
    const float* __restrict__ addend, int ldadd,
    int K, int nblkN, int tplus1)
{
  extern __shared__ unsigned short sA[];          // [32][K+8]
  const int ldsp  = K + 8;
  const int mb    = blockIdx.x / nblkN;
  const int nb    = blockIdx.x - mb * nblkN;
  const int m0blk = mb * 32;
  const int n0blk = nb * 256;

  // ---- stage A panel into LDS with async copies (16B chunks) ----
  {
    const int chunks = K >> 3;                    // 8 bf16 per 16B chunk
    for (int idx = threadIdx.x; idx < 32 * chunks; idx += 256) {
      int r  = idx / chunks;
      int cc = idx - r * chunks;
      unsigned ldsOff = (unsigned)(unsigned long long)(&sA[r * ldsp + (cc << 3)]);
      const unsigned short* g = &A[(size_t)(m0blk + r) * lda + (cc << 3)];
      asm volatile("global_load_async_to_lds_b128 %0, %1, off"
                   :: "v"(ldsOff), "v"(g) : "memory");
    }
    asm volatile("s_wait_asynccnt 0x0" ::: "memory");
  }
  __syncthreads();

  const int wave = threadIdx.x >> 5;
  const int lane = threadIdx.x & 31;
  const int l15  = lane & 15;
  const int hi16 = lane >> 4;                     // 0/1 (K-half select)

  // A frags per ISA 16-bit A layout: lanes 0-15 K 0..7/16..23, lanes 16-31 K 8..15/24..31
  const unsigned short* arow0 = &sA[l15 * ldsp + (hi16 << 3)];          // rows 0..15
  const unsigned short* arow1 = &sA[(16 + l15) * ldsp + (hi16 << 3)];   // rows 16..31
  // B frags: lane n = l15, same K layout as A: lanes 0-15 K 0..7/16..23, lanes 16-31 K 8..15/24..31
  const int wn0 = n0blk + wave * 32 + l15;
  const int wn1 = wn0 + 16;
  const unsigned short* wrow0 = W + (size_t)wn0 * ldw + (hi16 << 3);
  const unsigned short* wrow1 = W + (size_t)wn1 * ldw + (hi16 << 3);

  v8f acc00 = {0.f,0.f,0.f,0.f,0.f,0.f,0.f,0.f};
  v8f acc01 = acc00, acc10 = acc00, acc11 = acc00;

  #pragma unroll 2
  for (int k0 = 0; k0 < K; k0 += 32) {
    __builtin_prefetch(wrow0 + k0 + 256, 0, 1);   // global_prefetch_b8
    __builtin_prefetch(wrow1 + k0 + 256, 0, 1);
    ABFrag a0, a1, b0, b1;
    a0.q[0] = *(const v4u*)(arow0 + k0);
    a0.q[1] = *(const v4u*)(arow0 + k0 + 16);
    a1.q[0] = *(const v4u*)(arow1 + k0);
    a1.q[1] = *(const v4u*)(arow1 + k0 + 16);
    b0.q[0] = *(const v4u*)(wrow0 + k0);
    b0.q[1] = *(const v4u*)(wrow0 + k0 + 16);
    b1.q[0] = *(const v4u*)(wrow1 + k0);
    b1.q[1] = *(const v4u*)(wrow1 + k0 + 16);
    acc00 = __builtin_amdgcn_wmma_f32_16x16x32_bf16(false, a0.v, false, b0.v, (short)0, acc00, false, false);
    acc10 = __builtin_amdgcn_wmma_f32_16x16x32_bf16(false, a1.v, false, b0.v, (short)0, acc10, false, false);
    acc01 = __builtin_amdgcn_wmma_f32_16x16x32_bf16(false, a0.v, false, b1.v, (short)0, acc01, false, false);
    acc11 = __builtin_amdgcn_wmma_f32_16x16x32_bf16(false, a1.v, false, b1.v, (short)0, acc11, false, false);
  }

  // ---- epilogue: C/D layout: vgpr r, lane -> row = r + 8*hi16, col = l15 ----
  float bs0 = 0.f, bs1 = 0.f;
  if (bias)  { bs0 += bias[wn0];  bs1 += bias[wn1]; }
  if (bias2) { bs0 += bias2[wn0]; bs1 += bias2[wn1]; }

  auto store_tile = [&](const v8f& accv, int mtile, int wn, float bsum) {
    #pragma unroll
    for (int r = 0; r < 8; ++r) {
      const int ar = m0blk + (mtile << 4) + (hi16 << 3) + r;   // logical A row
      float v = accv[r] + bsum;
      if (HAS_ADD) v += addend[(size_t)ar * ldadd + wn];
      size_t orow = OUT_MAP ? (size_t)(ar & 31) * T_ + (ar >> 5) + tplus1
                            : (size_t)ar;
      size_t oidx = orow * (size_t)ldo + wn;
      if (ACCUM) v += D[oidx];
      D[oidx] = v;
    }
  };
  store_tile(acc00, 0, wn0, bs0);
  store_tile(acc01, 0, wn1, bs1);
  store_tile(acc10, 1, wn0, bs0);
  store_tile(acc11, 1, wn1, bs1);
}

// =====================================================================
// K1: per-batch attention: q = h@Wh^T ; scores = v . tanh(enc_proj+q);
//     masked softmax; context = aw@enc ; emit u=[ctx|h], y[512:]=ctx (bf16)
// =====================================================================
__global__ __launch_bounds__(256) void k1_attention(
    const float* __restrict__ h, const float* __restrict__ attn_W,
    const float* __restrict__ v_w, const float* __restrict__ enc_proj,
    const float* __restrict__ enc, const int* __restrict__ mask,
    unsigned short* __restrict__ u_bf, unsigned short* __restrict__ y_bf)
{
  __shared__ float hb[H_];
  __shared__ float qb[H_];
  __shared__ float red[256];
  __shared__ float awb[S_];
  const int b   = blockIdx.x;
  const int tid = threadIdx.x;

  for (int j = tid; j < H_; j += 256) hb[j] = h[b * H_ + j];
  __syncthreads();

  // q[r] = sum_k h[k] * attn_W[r, k]  (first H columns of attn_W)
  for (int r = tid; r < H_; r += 256) {
    const float4* wr = (const float4*)(attn_W + (size_t)r * (H_ + EH_));
    float s = 0.f;
    #pragma unroll 4
    for (int k4 = 0; k4 < H_ / 4; ++k4) {
      float4 w = wr[k4];
      int k = k4 * 4;
      s += w.x * hb[k] + w.y * hb[k+1] + w.z * hb[k+2] + w.w * hb[k+3];
    }
    qb[r] = s;
  }
  __syncthreads();

  // scores: 2 threads per s, each over 256 h
  const int si   = tid >> 1;
  const int half = tid & 1;
  {
    const float* ep = enc_proj + ((size_t)(b * S_ + si)) * H_ + half * 256;
    const float* qh = qb + half * 256;
    const float* vv = v_w + half * 256;
    float p = 0.f;
    for (int hh = 0; hh < 256; ++hh) p += vv[hh] * tanhf(ep[hh] + qh[hh]);
    red[tid] = p;
  }
  __syncthreads();
  float sc = 0.f;
  if (tid < S_) {
    sc = red[2 * tid] + red[2 * tid + 1];
    if (mask[b * S_ + tid] == 0) sc = -1.0e9f;
  }
  __syncthreads();
  red[tid] = (tid < S_) ? sc : -1.0e30f;
  __syncthreads();
  for (int off = 64; off > 0; off >>= 1) {
    if (tid < off) red[tid] = fmaxf(red[tid], red[tid + off]);
    __syncthreads();
  }
  const float mx = red[0];
  __syncthreads();
  const float ex = (tid < S_) ? __expf(sc - mx) : 0.f;
  red[tid] = ex;
  __syncthreads();
  for (int off = 128; off > 0; off >>= 1) {
    if (tid < off) red[tid] += red[tid + off];
    __syncthreads();
  }
  const float inv = 1.f / red[0];
  if (tid < S_) awb[tid] = ex * inv;
  __syncthreads();

  // context[e] = sum_s aw[s] * enc[b,s,e]
  for (int e = tid; e < EH_; e += 256) {
    const float* eb = enc + (size_t)b * S_ * EH_ + e;
    float a = 0.f;
    #pragma unroll 4
    for (int s2 = 0; s2 < S_; ++s2) a += awb[s2] * eb[(size_t)s2 * EH_];
    unsigned short bv = f2bf(a);
    u_bf[b * KU_ + e]       = bv;   // u = [ctx | h]
    y_bf[b * KU_ + H_ + e]  = bv;   // y = [h_new | ctx]
  }
  for (int j = tid; j < H_; j += 256) u_bf[b * KU_ + EH_ + j] = f2bf(hb[j]);
}

// =====================================================================
// K3: LSTM pointwise: c' = sig(f)*c + sig(i)*tanh(g); h' = sig(o)*tanh(c')
// =====================================================================
__global__ __launch_bounds__(256) void k3_pointwise(
    const float* __restrict__ gates, float* __restrict__ h,
    float* __restrict__ c, unsigned short* __restrict__ y_bf)
{
  const int idx = blockIdx.x * 256 + threadIdx.x;    // B*H = 16384
  const int b = idx >> 9, j = idx & (H_ - 1);
  const float* g = gates + (size_t)b * G4_;
  const float gi = g[j], gf = g[H_ + j], gg = g[2 * H_ + j], go = g[3 * H_ + j];
  const float si = 1.f / (1.f + __expf(-gi));
  const float sf = 1.f / (1.f + __expf(-gf));
  const float so = 1.f / (1.f + __expf(-go));
  const float cn = sf * c[idx] + si * tanhf(gg);
  const float hn = so * tanhf(cn);
  c[idx] = cn;
  h[idx] = hn;
  y_bf[b * KU_ + j] = f2bf(hn);
}

// ---------------- precompute / conversion kernels ----------------
__global__ void k_conv_sub(const float* __restrict__ src, int srcld, int col0,
                           unsigned short* __restrict__ dst, long long n, int cols)
{
  for (long long i = (long long)blockIdx.x * 256 + threadIdx.x; i < n;
       i += (long long)gridDim.x * 256) {
    long long r = i / cols;
    int cc = (int)(i - r * cols);
    dst[i] = f2bf(src[r * (long long)srcld + col0 + cc]);
  }
}

// Wg[r, 0:1024] = W_ih[r, 256:1280] ; Wg[r, 1024:1536] = W_hh[r, :]
__global__ void k_make_wg(const float* __restrict__ W_ih,
                          const float* __restrict__ W_hh,
                          unsigned short* __restrict__ dst)
{
  for (int i = blockIdx.x * 256 + threadIdx.x; i < G4_ * KU_; i += gridDim.x * 256) {
    int r = i / KU_, j = i - r * KU_;
    float v = (j < EH_) ? W_ih[(size_t)r * (E_ + EH_) + E_ + j]
                        : W_hh[(size_t)r * H_ + (j - EH_)];
    dst[i] = f2bf(v);
  }
}

// emb_bf[t*B + b, e] = (tok==0) ? 0 : emb_table[tok, e]   (padding_idx = 0)
__global__ void k_embed(const int* __restrict__ tgt,
                        const float* __restrict__ tab,
                        unsigned short* __restrict__ dst)
{
  for (int i = blockIdx.x * 256 + threadIdx.x; i < TS_ * B_ * E_; i += gridDim.x * 256) {
    int e = i & (E_ - 1);
    int rb = i >> 8;                 // t*B + b
    int t = rb >> 5, b = rb & 31;
    int tok = tgt[b * T_ + t];
    dst[i] = f2bf(tok == 0 ? 0.f : tab[(size_t)tok * E_ + e]);
  }
}

__global__ void k_init(const float* __restrict__ hid, const float* __restrict__ cel,
                       float* __restrict__ h, float* __restrict__ c)
{
  int i = blockIdx.x * 256 + threadIdx.x;
  if (i < B_ * H_) { h[i] = hid[i]; c[i] = cel[i]; }
}

__global__ void k_zero0(float* __restrict__ out)
{
  for (int i = blockIdx.x * 256 + threadIdx.x; i < B_ * V_; i += gridDim.x * 256) {
    int b = i / V_, v = i - b * V_;
    out[(size_t)b * T_ * V_ + v] = 0.f;
  }
}

// =====================================================================
extern "C" void kernel_launch(void* const* d_in, const int* in_sizes, int n_in,
                              void* d_out, int out_size, void* d_ws, size_t ws_size,
                              hipStream_t stream)
{
  (void)in_sizes; (void)n_in; (void)out_size; (void)ws_size;
  const int*   tgt    = (const int*)  d_in[0];
  const float* hidden = (const float*)d_in[1];
  const float* cell   = (const float*)d_in[2];
  const float* enc    = (const float*)d_in[3];
  const int*   mask   = (const int*)  d_in[4];
  const float* emb_t  = (const float*)d_in[5];
  const float* attn_W = (const float*)d_in[6];
  const float* attn_b = (const float*)d_in[7];
  const float* v_w    = (const float*)d_in[8];
  const float* W_ih   = (const float*)d_in[9];
  const float* W_hh   = (const float*)d_in[10];
  const float* b_ih   = (const float*)d_in[11];
  const float* b_hh   = (const float*)d_in[12];
  const float* out_W  = (const float*)d_in[13];
  const float* out_b  = (const float*)d_in[14];
  float* out = (float*)d_out;

  // ---- workspace carve (total ~158 MB) ----
  char* p = (char*)d_ws;
  auto carve = [&](size_t bytes) {
    char* r = p; p += (bytes + 255) & ~(size_t)255; return (void*)r;
  };
  unsigned short* enc_bf   = (unsigned short*)carve((size_t)B_ * S_ * EH_ * 2);
  unsigned short* We_bf    = (unsigned short*)carve((size_t)H_ * EH_ * 2);
  unsigned short* Wg_bf    = (unsigned short*)carve((size_t)G4_ * KU_ * 2);
  unsigned short* Wout_bf  = (unsigned short*)carve((size_t)V_ * KU_ * 2);
  unsigned short* WoutE_bf = (unsigned short*)carve((size_t)V_ * E_ * 2);
  unsigned short* WihE_bf  = (unsigned short*)carve((size_t)G4_ * E_ * 2);
  unsigned short* emb_bf   = (unsigned short*)carve((size_t)TS_ * B_ * E_ * 2);
  float* enc_proj  = (float*)carve((size_t)B_ * S_ * H_ * 4);
  float* pre_gates = (float*)carve((size_t)TS_ * B_ * G4_ * 4);
  float* hbuf      = (float*)carve((size_t)B_ * H_ * 4);
  float* cbuf      = (float*)carve((size_t)B_ * H_ * 4);
  unsigned short* u_bf = (unsigned short*)carve((size_t)B_ * KU_ * 2);
  unsigned short* y_bf = (unsigned short*)carve((size_t)B_ * KU_ * 2);
  float* gates     = (float*)carve((size_t)B_ * G4_ * 4);

  auto shA = [](int K) { return (size_t)32 * (K + 8) * 2; };

  // ---- one-time conversions / precompute (parallel) ----
  k_conv_sub<<<2048, 256, 0, stream>>>(enc, EH_, 0, enc_bf, (long long)B_ * S_ * EH_, EH_);
  k_conv_sub<<<512,  256, 0, stream>>>(attn_W, H_ + EH_, H_, We_bf, (long long)H_ * EH_, EH_);
  k_conv_sub<<<4096, 256, 0, stream>>>(out_W, H_ + EH_ + E_, 0, Wout_bf, (long long)V_ * KU_, KU_);
  k_conv_sub<<<2048, 256, 0, stream>>>(out_W, H_ + EH_ + E_, KU_, WoutE_bf, (long long)V_ * E_, E_);
  k_conv_sub<<<512,  256, 0, stream>>>(W_ih, E_ + EH_, 0, WihE_bf, (long long)G4_ * E_, E_);
  k_make_wg<<<2048, 256, 0, stream>>>(W_ih, W_hh, Wg_bf);
  k_embed<<<512, 256, 0, stream>>>(tgt, emb_t, emb_bf);
  k_init<<<(B_ * H_ + 255) / 256, 256, 0, stream>>>(hidden, cell, hbuf, cbuf);
  k_zero0<<<2048, 256, 0, stream>>>(out);

  // enc_proj[BS, H] = enc_bf @ We^T + attn_b            (M=4096, N=512, K=1024)
  gemm_wmma<false, false, false><<<2 * 128, 256, shA(EH_), stream>>>(
      enc_bf, EH_, We_bf, EH_, enc_proj, H_, attn_b, nullptr, nullptr, 0, EH_, 2, 0);
  // pre_gates[TS*B, 4H] = emb @ W_ih[:, :E]^T + b_ih + b_hh   (M=2016, N=2048, K=256)
  gemm_wmma<false, false, false><<<8 * 63, 256, shA(E_), stream>>>(
      emb_bf, E_, WihE_bf, E_, pre_gates, G4_, b_ih, b_hh, nullptr, 0, E_, 8, 0);
  // d_out[b, t+1, :] = emb @ out_W[:, 1536:]^T + out_b        (M=2016, N=32000, K=256)
  gemm_wmma<true, false, false><<<125 * 63, 256, shA(E_), stream>>>(
      emb_bf, E_, WoutE_bf, E_, out, V_, out_b, nullptr, nullptr, 0, E_, 125, 1);

  // ---- serial recurrence: 63 steps ----
  for (int t = 0; t < TS_; ++t) {
    k1_attention<<<B_, 256, 0, stream>>>(hbuf, attn_W, v_w, enc_proj, enc, mask, u_bf, y_bf);
    // gates = [ctx|h] @ Wg^T + pre_gates[t]                    (M=32, N=2048, K=1536)
    gemm_wmma<false, false, true><<<8, 256, shA(KU_), stream>>>(
        u_bf, KU_, Wg_bf, KU_, gates, G4_, nullptr, nullptr,
        pre_gates + (size_t)t * B_ * G4_, G4_, KU_, 8, 0);
    k3_pointwise<<<(B_ * H_) / 256, 256, 0, stream>>>(gates, hbuf, cbuf, y_bf);
    // d_out[b, t+1, :] += [h'|ctx] @ out_W[:, :1536]^T         (M=32, N=32000, K=1536)
    gemm_wmma<true, true, false><<<125, 256, shA(KU_), stream>>>(
        y_bf, KU_, Wout_bf, KU_, out, V_, nullptr, nullptr, nullptr, 0, KU_, 125, t + 1);
  }
}
